// GlobalSelfAttention_29025388986371
// MI455X (gfx1250) — matrix operand order repaired and checked by reading, not compiled, not run on hardware
//
#include <hip/hip_runtime.h>
#include <stdint.h>

// ---------------------------------------------------------------------------
// Types / helpers
// ---------------------------------------------------------------------------
typedef __attribute__((ext_vector_type(16))) __bf16 bf16x16;
typedef __attribute__((ext_vector_type(8)))  float  f32x8;

union FragAB {
  bf16x16 v;
  uint4   q[2];
  unsigned short u[16];
};

__device__ __forceinline__ unsigned short f2bf(float f) {
  unsigned int u = __builtin_bit_cast(unsigned int, f);
  unsigned int r = (u + 0x7FFFu + ((u >> 16) & 1u)) >> 16;  // RNE
  return (unsigned short)r;
}
__device__ __forceinline__ float bf2f(unsigned short h) {
  unsigned int u = ((unsigned int)h) << 16;
  return __builtin_bit_cast(float, u);
}

#define WMMA_BF16(A, B, C) \
  __builtin_amdgcn_wmma_f32_16x16x32_bf16(false, (A), false, (B), (short)0, (C), false, false)

// Problem constants (match reference setup_inputs)
#define BB   8
#define NN   1024
#define CC   768
#define HH   12
#define DH   64
#define NIMG 784

// ---------------------------------------------------------------------------
// K0: fp32 -> bf16 conversion
// ---------------------------------------------------------------------------
__global__ void cvt_bf16_kernel(const float* __restrict__ src,
                                unsigned short* __restrict__ dst, int n) {
  int i = blockIdx.x * blockDim.x + threadIdx.x;
  if (i < n) dst[i] = f2bf(src[i]);
}

// ---------------------------------------------------------------------------
// K1: QKV GEMM.  X[8192,768] (bf16, row-major) x W[2304,768]^T (bf16, row-major)
// Each wave: 16x64 output strip.  Block: 8 waves -> 64x128 tile.
// Epilogue: +bias, scatter bf16 into Q/K/V [B,H,N,64].
// ---------------------------------------------------------------------------
__global__ __launch_bounds__(256) void qkv_gemm_kernel(
    const unsigned short* __restrict__ X,
    const unsigned short* __restrict__ W,
    const float* __restrict__ bias,
    unsigned short* __restrict__ Qo,
    unsigned short* __restrict__ Ko,
    unsigned short* __restrict__ Vo) {
  const int tid  = threadIdx.x;
  const int wv   = tid >> 5, lane = tid & 31;
  const int l    = lane & 15, hi = lane >> 4;
  const int wm   = wv & 3, wn = wv >> 2;
  const int mRow = blockIdx.x * 64 + wm * 16;
  const int nBase= blockIdx.y * 128 + wn * 64;

  f32x8 acc[4];
#pragma unroll
  for (int s4 = 0; s4 < 4; ++s4) acc[s4] = (f32x8){};

  const unsigned short* ap = X + (mRow + l) * CC;
  for (int k0 = 0; k0 < CC; k0 += 32) {
    FragAB a;
    a.q[0] = *(const uint4*)(ap + k0 + hi * 8);
    a.q[1] = *(const uint4*)(ap + k0 + 16 + hi * 8);
#pragma unroll
    for (int s4 = 0; s4 < 4; ++s4) {
      FragAB b;
      const unsigned short* bp = W + (nBase + s4 * 16 + l) * CC + k0 + hi * 8;
      b.q[0] = *(const uint4*)(bp);
      b.q[1] = *(const uint4*)(bp + 16);
      acc[s4] = WMMA_BF16(a.v, b.v, acc[s4]);
    }
  }

#pragma unroll
  for (int s4 = 0; s4 < 4; ++s4) {
    const int j   = nBase + s4 * 16 + l;     // column in [0,2304)
    const int sec = j / CC;                  // 0:q 1:k 2:v
    const int rem = j - sec * CC;
    const int h   = rem >> 6;
    const int d   = rem & 63;
    unsigned short* dst = (sec == 0) ? Qo : (sec == 1) ? Ko : Vo;
    const float bj = bias[j];
#pragma unroll
    for (int r = 0; r < 8; ++r) {
      const int mtok = mRow + r + 8 * hi;    // token index in [0, B*N)
      const int b    = mtok >> 10;
      const int n    = mtok & 1023;
      dst[((b * HH + h) * NN + n) * DH + d] = f2bf(acc[s4][r] + bj);
    }
  }
}

// ---------------------------------------------------------------------------
// K2: RoPE2D in-place on a [B,H,N,64] bf16 tensor (image-token prefix only).
// Thread handles one (b,h,n,half,i) pair: dims (half*32+i, half*32+i+16).
// ---------------------------------------------------------------------------
__global__ void rope2d_kernel(unsigned short* __restrict__ T,
                              const int* __restrict__ pos2d) {
  const int total = BB * HH * NIMG * 2 * 16;
  int idx = blockIdx.x * blockDim.x + threadIdx.x;
  if (idx >= total) return;
  const int i    = idx & 15;
  const int half = (idx >> 4) & 1;           // 0: y-half, 1: x-half
  const int n    = (idx >> 5) % NIMG;
  const int hb   = idx / (32 * NIMG);
  const int h    = hb % HH;
  const int b    = hb / HH;

  const int pos  = pos2d[(b * NIMG + n) * 2 + half];  // [...,0]=y, [...,1]=x
  const float freq = __powf(100.0f, -(float)i / 16.0f);
  const float ang  = (float)pos * freq;
  const float c = __cosf(ang), s = __sinf(ang);

  unsigned short* base = T + ((b * HH + h) * NN + n) * DH + half * 32 + i;
  const float v0 = bf2f(base[0]);
  const float v1 = bf2f(base[16]);
  base[0]  = f2bf(v0 * c - v1 * s);
  base[16] = f2bf(v1 * c + v0 * s);
}

// ---------------------------------------------------------------------------
// K3: fused flash attention.  Grid: (qblock=8, h=12, b=8), block = 256 (8 waves).
// Wave owns 16 query rows; iterates 16 key-blocks of 64, staging K/V in LDS.
// ---------------------------------------------------------------------------
__global__ __launch_bounds__(256) void attn_kernel(
    const unsigned short* __restrict__ Qb,
    const unsigned short* __restrict__ Kb,
    const unsigned short* __restrict__ Vb,
    unsigned short* __restrict__ Ctx) {      // [B,N,768] bf16
  __shared__ unsigned short Ks[64 * 64];     // [key][d]
  __shared__ unsigned short Vt[64 * 64];     // [d][key]  (transposed)
  __shared__ unsigned short Ps[8 * 16 * 64]; // per-wave P re-layout buffer

  const int qb = blockIdx.x, h = blockIdx.y, b = blockIdx.z;
  const int tid = threadIdx.x, wv = tid >> 5, lane = tid & 31;
  const int l = lane & 15, hi = lane >> 4;
  const int qRow0 = qb * 128 + wv * 16;

  const unsigned short* qp = Qb + (size_t)(b * HH + h) * NN * DH;
  const unsigned short* kp = Kb + (size_t)(b * HH + h) * NN * DH;
  const unsigned short* vp = Vb + (size_t)(b * HH + h) * NN * DH;

  // Q fragments (A-matrix, Kdim = head dim, two chunks of 32)
  FragAB qa0, qa1;
  {
    const unsigned short* p = qp + (qRow0 + l) * DH;
    qa0.q[0] = *(const uint4*)(p + hi * 8);
    qa0.q[1] = *(const uint4*)(p + 16 + hi * 8);
    qa1.q[0] = *(const uint4*)(p + 32 + hi * 8);
    qa1.q[1] = *(const uint4*)(p + 48 + hi * 8);
  }

  float m_r[8], l_r[8];
  f32x8 o[4];
#pragma unroll
  for (int r = 0; r < 8; ++r) { m_r[r] = -3.0e38f; l_r[r] = 0.0f; }
#pragma unroll
  for (int n = 0; n < 4; ++n) o[n] = (f32x8){};

  for (int kb = 0; kb < NN / 64; ++kb) {
    __syncthreads();  // previous iteration's LDS reads done
    {
      // K block: straight 8KB copy (512 x uint4)
      const uint4* src = (const uint4*)(kp + kb * 64 * DH);
      uint4* dst = (uint4*)Ks;
      dst[tid]       = src[tid];
      dst[tid + 256] = src[tid + 256];
      // V block: transpose into Vt[d][key]
      const unsigned short* vs = vp + kb * 64 * DH;
#pragma unroll
      for (int it = 0; it < 16; ++it) {
        const int idx = tid * 16 + it;       // 0..4095
        const int key = idx >> 6, d = idx & 63;
        Vt[d * 64 + key] = vs[idx];
      }
    }
    __syncthreads();

    // S = Q K^T : 4 tiles of 16x16, Kdim = 64 (2 WMMAs each)
    f32x8 s[4];
#pragma unroll
    for (int t4 = 0; t4 < 4; ++t4) {
      FragAB kb0, kb1;
      const unsigned short* p = Ks + (t4 * 16 + l) * 64;
      kb0.q[0] = *(const uint4*)(p + hi * 8);
      kb0.q[1] = *(const uint4*)(p + 16 + hi * 8);
      kb1.q[0] = *(const uint4*)(p + 32 + hi * 8);
      kb1.q[1] = *(const uint4*)(p + 48 + hi * 8);
      f32x8 z = (f32x8){};
      z = WMMA_BF16(qa0.v, kb0.v, z);
      s[t4] = WMMA_BF16(qa1.v, kb1.v, z);
    }

    // online softmax over this 64-key block
    unsigned short* pw = Ps + wv * (16 * 64);
#pragma unroll
    for (int r = 0; r < 8; ++r) {
      float mx = -3.0e38f;
#pragma unroll
      for (int t4 = 0; t4 < 4; ++t4) mx = fmaxf(mx, s[t4][r] * 0.125f);
      mx = fmaxf(mx, __shfl_xor(mx, 1));
      mx = fmaxf(mx, __shfl_xor(mx, 2));
      mx = fmaxf(mx, __shfl_xor(mx, 4));
      mx = fmaxf(mx, __shfl_xor(mx, 8));
      const float mnew = fmaxf(m_r[r], mx);
      const float corr = __expf(m_r[r] - mnew);
      float rs = 0.0f;
#pragma unroll
      for (int t4 = 0; t4 < 4; ++t4) {
        const float pv = __expf(s[t4][r] * 0.125f - mnew);
        rs += pv;
        pw[(r + 8 * hi) * 64 + t4 * 16 + l] = f2bf(pv);  // D-layout -> LDS
      }
      rs += __shfl_xor(rs, 1);
      rs += __shfl_xor(rs, 2);
      rs += __shfl_xor(rs, 4);
      rs += __shfl_xor(rs, 8);
      l_r[r] = l_r[r] * corr + rs;
      m_r[r] = mnew;
#pragma unroll
      for (int n = 0; n < 4; ++n) o[n][r] *= corr;
    }

    // reload P as A-matrix fragments (Kdim = 64 keys, two chunks of 32)
    FragAB pa0, pa1;
    {
      const unsigned short* p = pw + l * 64;
      pa0.q[0] = *(const uint4*)(p + hi * 8);
      pa0.q[1] = *(const uint4*)(p + 16 + hi * 8);
      pa1.q[0] = *(const uint4*)(p + 32 + hi * 8);
      pa1.q[1] = *(const uint4*)(p + 48 + hi * 8);
    }
    // O += P V : 4 dim-tiles of 16 (2 WMMAs each)
#pragma unroll
    for (int n = 0; n < 4; ++n) {
      FragAB vb0, vb1;
      const unsigned short* p = Vt + (n * 16 + l) * 64;
      vb0.q[0] = *(const uint4*)(p + hi * 8);
      vb0.q[1] = *(const uint4*)(p + 16 + hi * 8);
      vb1.q[0] = *(const uint4*)(p + 32 + hi * 8);
      vb1.q[1] = *(const uint4*)(p + 48 + hi * 8);
      o[n] = WMMA_BF16(pa0.v, vb0.v, o[n]);
      o[n] = WMMA_BF16(pa1.v, vb1.v, o[n]);
    }
  }

  // normalize and store context [B,N,768] bf16
#pragma unroll
  for (int n = 0; n < 4; ++n) {
#pragma unroll
    for (int r = 0; r < 8; ++r) {
      const int tok = qRow0 + r + 8 * hi;
      const float val = o[n][r] / l_r[r];
      Ctx[((size_t)(b * NN + tok)) * CC + h * DH + n * 16 + l] = f2bf(val);
    }
  }
}

// ---------------------------------------------------------------------------
// K4: output projection.  Ctx[8192,768] (bf16) x Wp[768,768]^T + b -> f32 out
// ---------------------------------------------------------------------------
__global__ __launch_bounds__(256) void proj_gemm_kernel(
    const unsigned short* __restrict__ X,
    const unsigned short* __restrict__ W,
    const float* __restrict__ bias,
    float* __restrict__ Out) {
  const int tid  = threadIdx.x;
  const int wv   = tid >> 5, lane = tid & 31;
  const int l    = lane & 15, hi = lane >> 4;
  const int wm   = wv & 3, wn = wv >> 2;
  const int mRow = blockIdx.x * 64 + wm * 16;
  const int nBase= blockIdx.y * 128 + wn * 64;

  f32x8 acc[4];
#pragma unroll
  for (int s4 = 0; s4 < 4; ++s4) acc[s4] = (f32x8){};

  const unsigned short* ap = X + (mRow + l) * CC;
  for (int k0 = 0; k0 < CC; k0 += 32) {
    FragAB a;
    a.q[0] = *(const uint4*)(ap + k0 + hi * 8);
    a.q[1] = *(const uint4*)(ap + k0 + 16 + hi * 8);
#pragma unroll
    for (int s4 = 0; s4 < 4; ++s4) {
      FragAB b;
      const unsigned short* bp = W + (nBase + s4 * 16 + l) * CC + k0 + hi * 8;
      b.q[0] = *(const uint4*)(bp);
      b.q[1] = *(const uint4*)(bp + 16);
      acc[s4] = WMMA_BF16(a.v, b.v, acc[s4]);
    }
  }

#pragma unroll
  for (int s4 = 0; s4 < 4; ++s4) {
    const int j = nBase + s4 * 16 + l;
    const float bj = bias[j];
#pragma unroll
    for (int r = 0; r < 8; ++r) {
      const int mtok = mRow + r + 8 * hi;
      Out[(size_t)mtok * CC + j] = acc[s4][r] + bj;
    }
  }
}

// ---------------------------------------------------------------------------
// Host launcher
// ---------------------------------------------------------------------------
extern "C" void kernel_launch(void* const* d_in, const int* in_sizes, int n_in,
                              void* d_out, int out_size, void* d_ws, size_t ws_size,
                              hipStream_t stream) {
  const float* x_t    = (const float*)d_in[0];   // [8,1024,768]
  const float* qkv_w  = (const float*)d_in[1];   // [2304,768]
  const float* qkv_b  = (const float*)d_in[2];   // [2304]
  const float* proj_w = (const float*)d_in[3];   // [768,768]
  const float* proj_b = (const float*)d_in[4];   // [768]
  const int*   pos2d  = (const int*)d_in[5];     // [8,784,2]
  float* out = (float*)d_out;

  char* ws = (char*)d_ws;
  size_t off = 0;
  unsigned short* xb    = (unsigned short*)(ws + off); off += (size_t)BB*NN*CC*2;       // 12.6MB
  unsigned short* wqkvb = (unsigned short*)(ws + off); off += (size_t)3*CC*CC*2;        // 3.5MB
  unsigned short* wprjb = (unsigned short*)(ws + off); off += (size_t)CC*CC*2;          // 1.2MB
  unsigned short* Qd    = (unsigned short*)(ws + off); off += (size_t)BB*HH*NN*DH*2;
  unsigned short* Kd    = (unsigned short*)(ws + off); off += (size_t)BB*HH*NN*DH*2;
  unsigned short* Vd    = (unsigned short*)(ws + off); off += (size_t)BB*HH*NN*DH*2;
  unsigned short* Ctx   = (unsigned short*)(ws + off); off += (size_t)BB*NN*CC*2;

  // K0: bf16 conversions
  {
    int n0 = BB * NN * CC;        // 6291456
    int n1 = 3 * CC * CC;         // 1769472
    int n2 = CC * CC;             // 589824
    cvt_bf16_kernel<<<(n0 + 255) / 256, 256, 0, stream>>>(x_t,    xb,    n0);
    cvt_bf16_kernel<<<(n1 + 255) / 256, 256, 0, stream>>>(qkv_w,  wqkvb, n1);
    cvt_bf16_kernel<<<(n2 + 255) / 256, 256, 0, stream>>>(proj_w, wprjb, n2);
  }

  // K1: QKV projection (M=8192, N=2304, K=768)
  qkv_gemm_kernel<<<dim3(8192 / 64, 2304 / 128), 256, 0, stream>>>(
      xb, wqkvb, qkv_b, Qd, Kd, Vd);

  // K2: RoPE2D on Q and K image-token prefix
  {
    int n = BB * HH * NIMG * 2 * 16;
    rope2d_kernel<<<(n + 255) / 256, 256, 0, stream>>>(Qd, pos2d);
    rope2d_kernel<<<(n + 255) / 256, 256, 0, stream>>>(Kd, pos2d);
  }

  // K3: fused attention
  attn_kernel<<<dim3(NN / 128, HH, BB), 256, 0, stream>>>(Qd, Kd, Vd, Ctx);

  // K4: output projection (M=8192, N=768, K=768)
  proj_gemm_kernel<<<dim3(8192 / 64, 768 / 128), 256, 0, stream>>>(
      Ctx, wprjb, proj_b, out);
}
